// EMAQuantizer_10024453669315
// MI455X (gfx1250) — matrix operand order repaired and checked by reading, not compiled, not run on hardware
//
#include <hip/hip_runtime.h>
#include <hip/hip_bf16.h>
#include <cstdint>

// ---------------------------------------------------------------------------
// VQ (EMA) quantizer forward for MI455X / gfx1250.
//
// inputs:     [4, 128, 32, 32, 32] f32   (B, C, D, H, W), C = embedding dim
// emb_weight: [1024, 128] f32            (K codes x C)
//
// d_out (float, flat, in return order):
//   [0 .. 16777216)            quantized_st  (channel-first, == x + (q - x))
//   [16777216]                 loss = 0.25 * mean((q - x)^2)
//   [16777217 .. +131072)      encoding_indices (as float values)
//   [.. +256)                  encodings_sum (zeros)
//   [.. +131072)               emb_weight passthrough
//
// Matrix path: argmin_k ( ||e_k||^2 - 2 x.e_k ) via bf16x3-split WMMA
// (hi*hi + hi*lo + lo*hi) on V_WMMA_F32_16X16X32_BF16. Each wave carries TWO
// 16-row M-tiles so every B fragment (code tile) feeds 6 WMMAs instead of 3:
// 24 WMMAs per 17 VMEM loads -> matrix-issue-bound instead of VMEM-bound.
// ---------------------------------------------------------------------------

typedef __attribute__((ext_vector_type(16))) __bf16 v16bf;
typedef __attribute__((ext_vector_type(8)))  __bf16 v8bf;
typedef __attribute__((ext_vector_type(8)))  float  v8f;

#define C_DIM        128
#define K_CODES      1024
#define SPATIAL      32768          // 32*32*32
#define N_ROWS       131072         // 4 * SPATIAL
#define ROWS_PER_BLK 128
#define THREADS      128            // 4 waves, each owns 2 M-tiles (32 rows)
#define LDS_STRIDE   136            // 128 + 8 pad (bf16 elems): 272B row stride
                                    // -> 4-bank shift per row, conflict-free b128

#define OFF_Q    ((size_t)0)
#define OFF_LOSS ((size_t)16777216)
#define OFF_IDX  ((size_t)16777217)
#define OFF_ESUM ((size_t)16908289)
#define OFF_EMB  ((size_t)16908545)

#define LOSS_SCALE (0.25f / 16777216.0f)   // commitment_cost / numel(inputs)

union AB { v16bf v; v8bf h[2]; };

__device__ __forceinline__ __bf16 f32_to_bf16(float f) {
  union { float f; uint32_t u; } a; a.f = f;
  uint32_t r = a.u + 0x7FFFu + ((a.u >> 16) & 1u);   // round-nearest-even
  union { uint16_t s; __bf16 b; } o; o.s = (uint16_t)(r >> 16);
  return o.b;
}
__device__ __forceinline__ float bf16_to_f32(__bf16 b) {
  union { uint16_t s; __bf16 b; } i; i.b = b;
  union { uint32_t u; float f; } o; o.u = ((uint32_t)i.s) << 16;
  return o.f;
}

// ---------------------------------------------------------------------------
// Prep: copy emb to output, split emb into bf16 hi/lo (row-major [K][C]),
// compute ||e_k||^2, zero the loss slot and encodings_sum.
// ---------------------------------------------------------------------------
__global__ void vq_prep(const float* __restrict__ emb,
                        float* __restrict__ out,
                        __bf16* __restrict__ ehi,
                        __bf16* __restrict__ elo,
                        float* __restrict__ esq) {
  int t = blockIdx.x * blockDim.x + threadIdx.x;
  if (t < K_CODES * C_DIM) {
    float v = emb[t];
    out[OFF_EMB + t] = v;
    __bf16 h = f32_to_bf16(v);
    __bf16 l = f32_to_bf16(v - bf16_to_f32(h));
    ehi[t] = h;
    elo[t] = l;
  }
  if (t < 256) out[OFF_ESUM + t] = 0.0f;
  if (t == 0)  out[OFF_LOSS] = 0.0f;
  if (t < K_CODES) {
    const float* row = emb + t * C_DIM;
    float s = 0.0f;
    #pragma unroll 4
    for (int c = 0; c < C_DIM; ++c) { float v = row[c]; s += v * v; }
    esq[t] = s;
  }
}

// ---------------------------------------------------------------------------
// Main: one block = 128 rows (all share batch b since 32768 % 128 == 0).
// 4 waves; wave w owns rows [w*32, w*32+32) as two 16-row M-tiles that share
// every B fragment. Per N-tile: 4 k-chunks x (2 tiles x 3 split-WMMAs) = 24.
// ---------------------------------------------------------------------------
__global__ void __launch_bounds__(THREADS, 1)
vq_main(const float* __restrict__ in,
        const float* __restrict__ emb,
        const __bf16* __restrict__ ehi,
        const __bf16* __restrict__ elo,
        const float* __restrict__ esq,
        float* __restrict__ out) {

  __shared__ alignas(16) __bf16 sAhi[ROWS_PER_BLK * LDS_STRIDE];
  __shared__ alignas(16) __bf16 sAlo[ROWS_PER_BLK * LDS_STRIDE];
  __shared__ int sIdx[ROWS_PER_BLK];

  const int tid  = threadIdx.x;
  const int lane = tid & 31;
  const int wave = tid >> 5;

  const int rowBase = blockIdx.x * ROWS_PER_BLK;
  const int b  = rowBase / SPATIAL;
  const int s0 = rowBase % SPATIAL;
  const float* xBase = in + (size_t)b * C_DIM * SPATIAL + s0;

  // ---- Stage X tile -> LDS as split bf16, layout [row][channel] (pad 136).
  // flat f = c*128 + j : consecutive threads take consecutive spatial j
  // -> fully coalesced global reads (channel stride is 32768 floats).
  for (int f = tid; f < ROWS_PER_BLK * C_DIM; f += THREADS) {
    int c = f >> 7;
    int j = f & 127;
    float v = xBase[(size_t)c * SPATIAL + j];
    __bf16 h = f32_to_bf16(v);
    __bf16 l = f32_to_bf16(v - bf16_to_f32(h));
    sAhi[j * LDS_STRIDE + c] = h;
    sAlo[j * LDS_STRIDE + c] = l;
  }
  __syncthreads();

  // ---- A fragments (16x32 bf16 per k-chunk), kept in VGPRs for the N loop.
  // ISA layout: lane L<16: row M=L,    elems 0..7 -> K=kb+0..7,  8..15 -> K=kb+16..23
  //             lane L>=16: row M=L-16, elems 0..7 -> K=kb+8..15, 8..15 -> K=kb+24..31
  const int mRow0 = wave * 32 + (lane & 15);       // M-tile 0
  const int mRow1 = mRow0 + 16;                    // M-tile 1
  const int hb    = (lane < 16) ? 0 : 8;
  const __bf16* aHi0 = sAhi + mRow0 * LDS_STRIDE;
  const __bf16* aLo0 = sAlo + mRow0 * LDS_STRIDE;
  const __bf16* aHi1 = sAhi + mRow1 * LDS_STRIDE;
  const __bf16* aLo1 = sAlo + mRow1 * LDS_STRIDE;
  AB fHi0[4], fLo0[4], fHi1[4], fLo1[4];
  #pragma unroll
  for (int kc = 0; kc < 4; ++kc) {
    int kb = kc * 32;
    fHi0[kc].h[0] = *(const v8bf*)(aHi0 + kb + hb);
    fHi0[kc].h[1] = *(const v8bf*)(aHi0 + kb + 16 + hb);
    fLo0[kc].h[0] = *(const v8bf*)(aLo0 + kb + hb);
    fLo0[kc].h[1] = *(const v8bf*)(aLo0 + kb + 16 + hb);
    fHi1[kc].h[0] = *(const v8bf*)(aHi1 + kb + hb);
    fHi1[kc].h[1] = *(const v8bf*)(aHi1 + kb + 16 + hb);
    fLo1[kc].h[0] = *(const v8bf*)(aLo1 + kb + hb);
    fLo1[kc].h[1] = *(const v8bf*)(aLo1 + kb + 16 + hb);
  }

  // ---- Running min over codes, per M-tile. C/D layout: VGPR r, lane L:
  //      M = r + 8*(L/16), N = L%16 -> lane tracks codes with n == L%16 (mod 16).
  float best0[8], best1[8];
  int   bidx0[8], bidx1[8];
  #pragma unroll
  for (int r = 0; r < 8; ++r) {
    best0[r] = 3.4e38f; bidx0[r] = 0x7fffffff;
    best1[r] = 3.4e38f; bidx1[r] = 0x7fffffff;
  }

  const int nCol  = lane & 15;
  const int kHalf = (lane < 16) ? 0 : 8;   // B 32x16: lanes 0-15 K=kb+0..7,16..23; 16-31 K=kb+8..15,24..31

  for (int tile = 0; tile < K_CODES / 16; ++tile) {
    const int code = tile * 16 + nCol;
    const __bf16* bHiP = ehi + code * C_DIM + kHalf;
    const __bf16* bLoP = elo + code * C_DIM + kHalf;

    v8f acc0 = {0.f, 0.f, 0.f, 0.f, 0.f, 0.f, 0.f, 0.f};
    v8f acc1 = {0.f, 0.f, 0.f, 0.f, 0.f, 0.f, 0.f, 0.f};
    #pragma unroll
    for (int kc = 0; kc < 4; ++kc) {
      AB bh, bl;
      bh.h[0] = *(const v8bf*)(bHiP + kc * 32);
      bh.h[1] = *(const v8bf*)(bHiP + kc * 32 + 16);
      bl.h[0] = *(const v8bf*)(bLoP + kc * 32);
      bl.h[1] = *(const v8bf*)(bLoP + kc * 32 + 16);
      // split product: hi*hi + hi*lo + lo*hi; B fragment reused by both M-tiles
      acc0 = __builtin_amdgcn_wmma_f32_16x16x32_bf16(false, fHi0[kc].v, false, bh.v,
                                                     (short)0, acc0, false, false);
      acc1 = __builtin_amdgcn_wmma_f32_16x16x32_bf16(false, fHi1[kc].v, false, bh.v,
                                                     (short)0, acc1, false, false);
      acc0 = __builtin_amdgcn_wmma_f32_16x16x32_bf16(false, fHi0[kc].v, false, bl.v,
                                                     (short)0, acc0, false, false);
      acc1 = __builtin_amdgcn_wmma_f32_16x16x32_bf16(false, fHi1[kc].v, false, bl.v,
                                                     (short)0, acc1, false, false);
      acc0 = __builtin_amdgcn_wmma_f32_16x16x32_bf16(false, fLo0[kc].v, false, bh.v,
                                                     (short)0, acc0, false, false);
      acc1 = __builtin_amdgcn_wmma_f32_16x16x32_bf16(false, fLo1[kc].v, false, bh.v,
                                                     (short)0, acc1, false, false);
    }

    const float eq = esq[code];
    #pragma unroll
    for (int r = 0; r < 8; ++r) {
      float d0 = eq - 2.0f * acc0[r];        // ||e||^2 - 2 x.e (x_sq constant/row)
      float d1 = eq - 2.0f * acc1[r];
      if (d0 < best0[r]) { best0[r] = d0; bidx0[r] = code; }  // first-occurrence
      if (d1 < best1[r]) { best1[r] = d1; bidx1[r] = code; }
    }
  }

  // ---- Cross-lane argmin within each 16-lane half (rows m=r resp. m=r+8).
  #pragma unroll
  for (int r = 0; r < 8; ++r) {
    float v0 = best0[r]; int i0 = bidx0[r];
    float v1 = best1[r]; int i1 = bidx1[r];
    #pragma unroll
    for (int m = 8; m >= 1; m >>= 1) {
      float ov0 = __shfl_xor(v0, m, 32);
      int   oi0 = __shfl_xor(i0, m, 32);
      float ov1 = __shfl_xor(v1, m, 32);
      int   oi1 = __shfl_xor(i1, m, 32);
      if (ov0 < v0 || (ov0 == v0 && oi0 < i0)) { v0 = ov0; i0 = oi0; }
      if (ov1 < v1 || (ov1 == v1 && oi1 < i1)) { v1 = ov1; i1 = oi1; }
    }
    bidx0[r] = i0; bidx1[r] = i1;
  }
  if ((lane & 15) == 0) {
    const int half = lane >> 4;
    #pragma unroll
    for (int r = 0; r < 8; ++r) {
      int m    = r + half * 8;
      int row0 = wave * 32 + m;
      int row1 = row0 + 16;
      sIdx[row0] = bidx0[r];
      sIdx[row1] = bidx1[r];
      out[OFF_IDX + (size_t)(rowBase + row0)] = (float)bidx0[r];
      out[OFF_IDX + (size_t)(rowBase + row1)] = (float)bidx1[r];
    }
  }
  __syncthreads();

  // ---- Gather fp32 embedding, write quantized_st (channel-first, coalesced:
  //      all 128 threads share channel c, consecutive spatial j), loss partial.
  float lsum = 0.0f;
  float* qBase = out + OFF_Q + (size_t)b * C_DIM * SPATIAL + s0;
  for (int f = tid; f < ROWS_PER_BLK * C_DIM; f += THREADS) {
    int c = f >> 7;
    int j = f & 127;
    int idx = sIdx[j];
    float x = xBase[(size_t)c * SPATIAL + j];
    float q = emb[idx * C_DIM + c];
    float diff = q - x;
    lsum += diff * diff;
    qBase[(size_t)c * SPATIAL + j] = x + diff;   // x + stopgrad(q - x)
  }
  #pragma unroll
  for (int m = 16; m >= 1; m >>= 1) lsum += __shfl_xor(lsum, m, 32);
  if (lane == 0) atomicAdd(out + OFF_LOSS, lsum * LOSS_SCALE);
}

// ---------------------------------------------------------------------------
extern "C" void kernel_launch(void* const* d_in, const int* in_sizes, int n_in,
                              void* d_out, int out_size, void* d_ws, size_t ws_size,
                              hipStream_t stream) {
  const float* d_inputs = (const float*)d_in[0];   // [4,128,32,32,32] f32
  const float* d_emb    = (const float*)d_in[1];   // [1024,128] f32
  float* out = (float*)d_out;

  // workspace: Ehi (256KB) | Elo (256KB) | esq (4KB)
  char* ws = (char*)d_ws;
  __bf16* ehi = (__bf16*)ws;
  __bf16* elo = (__bf16*)(ws + 262144);
  float*  wesq = (float*)(ws + 524288);

  vq_prep<<<512, 256, 0, stream>>>(d_emb, out, ehi, elo, wesq);
  vq_main<<<N_ROWS / ROWS_PER_BLK, THREADS, 0, stream>>>(d_inputs, d_emb, ehi, elo,
                                                         wesq, out);
  (void)in_sizes; (void)n_in; (void)out_size; (void)ws_size;
}
